// TopKSAE_53618371723772
// MI455X (gfx1250) — matrix operand order repaired and checked by reading, not compiled, not run on hardware
//
#include <hip/hip_runtime.h>

// ---------------------------------------------------------------------------
// TopK SAE forward for MI455X (gfx1250, wave32, WMMA).
//   Phase 1: z = x @ W^T + b  via v_wmma_f32_16x16x32_bf16 with split-bf16
//            (hi/lo) 3-term emulation for ~fp32 accuracy. Double-buffered
//            LDS pipeline: next k-slab's global loads fly under the WMMAs.
//            z stored non-temporally (keep L2 for the 75MB W working set).
//   Phase 2: per-row radix-select of the 64th largest value (row cached in
//            320KB-class LDS), then in-place mask to exactly K survivors.
// ---------------------------------------------------------------------------

#define D_IN   768
#define D_DICT 24576
#define TOPK   64
#define NROWS  8192

typedef __attribute__((ext_vector_type(16))) __bf16 v16bf;
typedef __attribute__((ext_vector_type(8)))  __bf16 v8bf;
typedef __attribute__((ext_vector_type(8)))  float  v8f;

// Order-preserving float -> uint key (ascending).
__device__ __forceinline__ unsigned fkey(float f) {
    unsigned u = __float_as_uint(f);
    return (u & 0x80000000u) ? ~u : (u | 0x80000000u);
}

// ---------------------------------------------------------------------------
// Phase 1: tiled GEMM. WG tile = 128(M) x 128(N), K-step 32.
// 256 threads = 8 wave32, wave grid 2(M) x 4(N): each wave owns 64x32
// = 4x2 WMMA 16x16 accumulator tiles.
// ---------------------------------------------------------------------------
__global__ __launch_bounds__(256) void sae_encode_gemm(
    const float* __restrict__ x,      // [NROWS][D_IN]
    const float* __restrict__ W,      // [D_DICT][D_IN]
    const float* __restrict__ bias,   // [D_DICT]
    float* __restrict__ z)            // [NROWS][D_DICT]
{
    // Double-buffered bf16 hi/lo tiles: 2 * 4 * 128*32*2B = 64 KB.
    __shared__ __align__(32) __bf16 Ahi[2][128][32];
    __shared__ __align__(32) __bf16 Alo[2][128][32];
    __shared__ __align__(32) __bf16 Bhi[2][128][32];
    __shared__ __align__(32) __bf16 Blo[2][128][32];

    const int t    = threadIdx.x;
    const int lane = t & 31;
    const int wave = t >> 5;
    const int wm   = wave & 1;     // 0..1 -> 64-row slab
    const int wn   = wave >> 1;    // 0..3 -> 32-col slab
    const int m0   = blockIdx.y * 128;
    const int n0   = blockIdx.x * 128;
    const int half = lane >> 4;    // hi/lo 16-lane group (K split)
    const int l16  = lane & 15;

    // Per-thread staging slots: 4 float4 of A + 4 float4 of B.
    const int sf_row = t >> 3;          // 0..31 base row (each i adds 32)
    const int sf_k0  = (t & 7) << 2;    // 0,4,..,28

    float4 ar[4], br[4];

    auto load_slab = [&](int kk) {
#pragma unroll
        for (int i = 0; i < 4; ++i) {
            const int row = sf_row + i * 32;
            ar[i] = *(const float4*)(x + (size_t)(m0 + row) * D_IN + kk + sf_k0);
            br[i] = *(const float4*)(W + (size_t)(n0 + row) * D_IN + kk + sf_k0);
        }
    };
    auto cvt_store_slab = [&](int buf) {
#pragma unroll
        for (int i = 0; i < 4; ++i) {
            const int row = sf_row + i * 32;
            const float va[4] = {ar[i].x, ar[i].y, ar[i].z, ar[i].w};
            const float vb[4] = {br[i].x, br[i].y, br[i].z, br[i].w};
#pragma unroll
            for (int j = 0; j < 4; ++j) {
                const __bf16 ha = (__bf16)va[j];
                Ahi[buf][row][sf_k0 + j] = ha;
                Alo[buf][row][sf_k0 + j] = (__bf16)(va[j] - (float)ha);
                const __bf16 hb = (__bf16)vb[j];
                Bhi[buf][row][sf_k0 + j] = hb;
                Blo[buf][row][sf_k0 + j] = (__bf16)(vb[j] - (float)hb);
            }
        }
    };

    v8f acc[4][2];
#pragma unroll
    for (int mt = 0; mt < 4; ++mt)
#pragma unroll
        for (int nt = 0; nt < 2; ++nt) acc[mt][nt] = (v8f)(0.0f);

    // Prologue: stage slab 0 into buffer 0.
    load_slab(0);
    cvt_store_slab(0);
    __syncthreads();

    int buf = 0;
    for (int kk = 0; kk < D_IN; kk += 32, buf ^= 1) {
        const bool more = (kk + 32) < D_IN;
        // Issue next slab's global loads now; they complete under the WMMAs.
        if (more) load_slab(kk + 32);

        // ---- load WMMA fragments from LDS buffer `buf` ----
        // A (16x32 bf16): lane l<16 holds row M=l, K={kb..kb+15}
        // with kb = 16*half.  B (32x16): lane holds col N=l16, K contiguous 16.
        v16bf ah[4], al[4], bh[2], bl[2];
#pragma unroll
        for (int mt = 0; mt < 4; ++mt) {
            const int r  = wm * 64 + mt * 16 + l16;
            const int kb = half * 16;
            v8bf p = *(const v8bf*)&Ahi[buf][r][kb];
            v8bf q = *(const v8bf*)&Ahi[buf][r][kb + 8];
            ah[mt] = __builtin_shufflevector(p, q, 0,1,2,3,4,5,6,7,8,9,10,11,12,13,14,15);
            p = *(const v8bf*)&Alo[buf][r][kb];
            q = *(const v8bf*)&Alo[buf][r][kb + 8];
            al[mt] = __builtin_shufflevector(p, q, 0,1,2,3,4,5,6,7,8,9,10,11,12,13,14,15);
        }
#pragma unroll
        for (int nt = 0; nt < 2; ++nt) {
            const int c  = wn * 32 + nt * 16 + l16;
            const int kb = half * 16;
            bh[nt] = *(const v16bf*)&Bhi[buf][c][kb];
            bl[nt] = *(const v16bf*)&Blo[buf][c][kb];
        }

        // ---- 3-term split-bf16 WMMA: ah*bh + ah*bl + al*bh ----
        // 8 independent accumulator tiles interleaved between chain links.
#pragma unroll
        for (int mt = 0; mt < 4; ++mt)
#pragma unroll
            for (int nt = 0; nt < 2; ++nt)
                acc[mt][nt] = __builtin_amdgcn_wmma_f32_16x16x32_bf16(
                    false, ah[mt], false, bh[nt], (short)0, acc[mt][nt], false, false);
#pragma unroll
        for (int mt = 0; mt < 4; ++mt)
#pragma unroll
            for (int nt = 0; nt < 2; ++nt)
                acc[mt][nt] = __builtin_amdgcn_wmma_f32_16x16x32_bf16(
                    false, ah[mt], false, bl[nt], (short)0, acc[mt][nt], false, false);
#pragma unroll
        for (int mt = 0; mt < 4; ++mt)
#pragma unroll
            for (int nt = 0; nt < 2; ++nt)
                acc[mt][nt] = __builtin_amdgcn_wmma_f32_16x16x32_bf16(
                    false, al[mt], false, bh[nt], (short)0, acc[mt][nt], false, false);

        // Convert + store the next slab into the other buffer, then one
        // barrier covers both "stores visible" and "reads of `buf` done
        // before it is overwritten next iteration".
        if (more) cvt_store_slab(buf ^ 1);
        __syncthreads();
    }

    // ---- epilogue: add bias, store non-temporally.  C layout: elem i ->
    // row = i + 8*half, col = l16 (within the 16x16 tile). ----
#pragma unroll
    for (int nt = 0; nt < 2; ++nt) {
        const int col  = n0 + wn * 32 + nt * 16 + l16;
        const float bb = bias[col];
#pragma unroll
        for (int mt = 0; mt < 4; ++mt) {
            const v8f c = acc[mt][nt];
#pragma unroll
            for (int i = 0; i < 8; ++i) {
                const int row = m0 + wm * 64 + mt * 16 + i + 8 * half;
                __builtin_nontemporal_store(c[i] + bb, &z[(size_t)row * D_DICT + col]);
            }
        }
    }
}

// ---------------------------------------------------------------------------
// Phase 2: one 256-thread WG per row.  Row (96KB) cached in LDS; 4x8-bit
// MSB-first radix select finds the key of the K-th largest; mask keeps
// exactly K entries (tie budget), written back in place.
// ---------------------------------------------------------------------------
__global__ __launch_bounds__(256) void sae_topk_mask(float* __restrict__ z)
{
    __shared__ float    rowv[D_DICT];     // 96 KB row cache (CDNA5: 320KB/WGP)
    __shared__ unsigned hist[256];
    __shared__ unsigned s_prefix, s_rem, s_tie;

    const int t = threadIdx.x;
    float* g = z + (size_t)blockIdx.x * D_DICT;

    // Each thread owns elements i == t (mod 256): no cross-thread rowv deps.
    for (int i = t; i < D_DICT; i += 256) rowv[i] = g[i];

    unsigned prefix = 0, rem = TOPK;
    for (int shift = 24; shift >= 0; shift -= 8) {
        hist[t] = 0;
        __syncthreads();
        const unsigned hm = (shift == 24) ? 0u : (0xFFFFFFFFu << (shift + 8));
        for (int i = t; i < D_DICT; i += 256) {
            const unsigned u = fkey(rowv[i]);
            if ((u & hm) == prefix) atomicAdd(&hist[(u >> shift) & 0xFFu], 1u);
        }
        __syncthreads();
        if (t == 0) {
            unsigned acc = 0; int sel = 0;
            for (int b = 255; b >= 0; --b) {
                const unsigned c = hist[b];
                if (acc + c >= rem) { sel = b; break; }
                acc += c;
            }
            s_prefix = prefix | ((unsigned)sel << shift);
            s_rem    = rem - acc;          // rank within the selected bin
        }
        __syncthreads();
        prefix = s_prefix;
        rem    = s_rem;
        __syncthreads();
    }
    const unsigned T      = prefix;        // exact key of K-th largest
    const unsigned budget = rem;           // how many ties at T to keep

    if (t == 0) s_tie = 0;
    __syncthreads();
    for (int i = t; i < D_DICT; i += 256) {
        const float v    = rowv[i];
        const unsigned u = fkey(v);
        float o = 0.0f;
        if (u > T)       o = v;
        else if (u == T) { if (atomicAdd(&s_tie, 1u) < budget) o = v; }
        g[i] = o;
    }
}

// ---------------------------------------------------------------------------
extern "C" void kernel_launch(void* const* d_in, const int* in_sizes, int n_in,
                              void* d_out, int out_size, void* d_ws, size_t ws_size,
                              hipStream_t stream) {
    (void)in_sizes; (void)n_in; (void)out_size; (void)d_ws; (void)ws_size;
    const float* x = (const float*)d_in[0];   // [8192, 768]
    const float* W = (const float*)d_in[1];   // [24576, 768]
    const float* b = (const float*)d_in[2];   // [24576]
    float* out = (float*)d_out;               // [8192, 24576]

    dim3 grid(D_DICT / 128, NROWS / 128);     // (192, 64)
    sae_encode_gemm<<<grid, 256, 0, stream>>>(x, W, b, out);
    sae_topk_mask<<<NROWS, 256, 0, stream>>>(out);
}
